// BranchContrastiveMarginLoss_37074157699124
// MI455X (gfx1250) — compile-verified, matrix operand order checked by reading, not run on hardware
//
#include <hip/hip_runtime.h>
#include <hip/hip_bf16.h>

// ---------------------------------------------------------------------------
// Hyperbolic (Poincare-ball) contrastive margin loss, CDNA5/gfx1250.
// N=32768 rows, D=32, G=8 groups x M=4096, NNEG=28672, K(top-k)=512.
// Negative term fused into v_wmma_f32_16x16x32_f16 Gram GEMM (D == WMMA K).
// Division-free trigger: arg < cosh(M) <=> sq < ((coshM-1)/2)*denom; the 8
// per-element tests fold into one min-reduction -> ONE branch per 16x16 tile.
// Slow path (rare) computes the exact clamped hinge max(MARGIN - d, 0), so
// the trigger may be loose without changing the result.
// ---------------------------------------------------------------------------

typedef __attribute__((ext_vector_type(16))) _Float16 v16h;
typedef __attribute__((ext_vector_type(8)))  float    v8f;

#define NROWS 32768
#define DIM   32
#define NGRP  8
#define GM    4096
#define NNEG  28672
#define EPSB  1e-5f
#define MAXN  (1.0f - EPSB)
#define MARGIN 0.02f
#define HALF_CM1    1.0000333335e-4f     // (cosh(0.02)-1)/2
#define SMALLC      1.0000333e-11f       // 1e-7 * HALF_CM1

// workspace layout (bytes)
#define WS_ACC    0                         // 4 floats: [0]=pos_sum [1]=hinge_sum
#define WS_PROJ   1024                      // N*D f32      (4 MiB)
#define WS_P2     (WS_PROJ + NROWS*DIM*4)   // N f32        (128 KiB)
#define WS_FH     (WS_P2   + NROWS*4)       // N*D f16      (2 MiB)
#define WS_CENT   (WS_FH   + NROWS*DIM*2)   // G*D f32
#define WS_C2     (WS_CENT + NGRP*DIM*4)    // G f32

union Frag { v16h v; uint4 q[2]; };

__device__ __forceinline__ float lane_sum32(float v) {
  #pragma unroll
  for (int o = 16; o > 0; o >>= 1) v += __shfl_xor(v, o, 32);
  return v;
}

// ---- init: zero the scalar accumulators ------------------------------------
__global__ void k_init(float* acc) {
  if (threadIdx.x < 4) acc[threadIdx.x] = 0.0f;
}

// ---- project rows to ball; emit f32 proj, f16 proj, squared norms ----------
__global__ void k_project(const float* __restrict__ emb,
                          float* __restrict__ proj,
                          float* __restrict__ p2,
                          _Float16* __restrict__ fh) {
  const int tid  = threadIdx.x;
  const int row  = blockIdx.x * 8 + (tid >> 5);
  const int d    = tid & 31;
  const float x  = emb[row * DIM + d];
  const float n2 = lane_sum32(x * x);
  const float n  = sqrtf(n2);
  const float sc = (n > MAXN) ? (MAXN / fmaxf(n, EPSB)) : 1.0f;
  const float p  = x * sc;
  proj[row * DIM + d] = p;
  fh[row * DIM + d]   = (_Float16)p;
  if (d == 0) p2[row] = n2 * sc * sc;
}

// ---- per-group projected centroid ------------------------------------------
__global__ void k_centroid(const float* __restrict__ proj,
                           float* __restrict__ cent,
                           float* __restrict__ c2) {
  const int g = blockIdx.x;
  const int tid = threadIdx.x;
  const int d = tid & 31, s = tid >> 5;           // 8 row-stripes x 32 dims
  __shared__ float part[8][32];
  float sum = 0.0f;
  for (int r = s; r < GM; r += 8) sum += proj[(g * GM + r) * DIM + d];
  part[s][d] = sum;
  __syncthreads();
  if (tid < 32) {
    float tot = 0.0f;
    #pragma unroll
    for (int ss = 0; ss < 8; ++ss) tot += part[ss][tid];
    const float mean = tot / (float)GM;
    const float n2 = lane_sum32(mean * mean);
    const float n  = sqrtf(n2);
    const float sc = (n > MAXN) ? (MAXN / fmaxf(n, EPSB)) : 1.0f;
    cent[g * DIM + tid] = mean * sc;
    if (tid == 0) c2[g] = n2 * sc * sc;
  }
}

// ---- positive term: sum of arccosh distances member -> centroid ------------
__global__ void k_positive(const float* __restrict__ proj,
                           const float* __restrict__ p2,
                           const float* __restrict__ cent,
                           const float* __restrict__ c2,
                           float* __restrict__ acc) {
  const int tid  = threadIdx.x;
  const int m    = blockIdx.x * 8 + (tid >> 5);   // global row
  const int g    = m / GM;
  const int d    = tid & 31;
  const float diff = proj[m * DIM + d] - cent[g * DIM + d];
  const float sq   = lane_sum32(diff * diff);
  __shared__ float sdata[8];
  if (d == 0) {
    const float denom = fmaxf((1.0f - p2[m]) * (1.0f - c2[g]), 1e-7f);
    const float arg   = 1.0f + 2.0f * sq / denom;
    sdata[tid >> 5]   = acoshf(fmaxf(arg, 1.0f + 1e-7f));
  }
  __syncthreads();
  if (tid == 0) {
    float t = 0.0f;
    #pragma unroll
    for (int i = 0; i < 8; ++i) t += sdata[i];
    atomicAdd(&acc[0], t);
  }
}

// ---- hinge over one 16x16 WMMA result tile ---------------------------------
// Fast path: 8x (fma, mul, max, sub, min) then ONE compare+branch per tile.
// Slow path recomputes all 8 elements with the exact clamped hinge.
__device__ __forceinline__ float hinge_tile(const v8f c, float n2,
                                            const float* m2v,
                                            const float* om2,
                                            const float* om2h) {
  const float on2 = 1.0f - n2;
  float s2v[8];
  float mind = 0.0f;
  #pragma unroll
  for (int r = 0; r < 8; ++r) {
    s2v[r] = fmaf(-2.0f, c[r], m2v[r] + n2);          // ||x-y||^2 (unclamped)
    const float thr = fmaxf(om2h[r] * on2, SMALLC);   // ((coshM-1)/2)*denom
    mind = fminf(mind, s2v[r] - thr);
  }
  float h = 0.0f;
  if (mind < 0.0f) {                                   // rare; one branch/tile
    #pragma unroll
    for (int r = 0; r < 8; ++r) {
      const float sq    = fmaxf(s2v[r], 0.0f);
      const float denom = fmaxf(om2[r] * on2, 1e-7f);
      const float arg   = 1.0f + 2.0f * sq / denom;
      const float dlt   = acoshf(fmaxf(arg, 1.0f + 1e-7f));
      h += fmaxf(MARGIN - dlt, 0.0f);                  // exact hinge (>=0)
    }
  }
  return h;
}

// ---- negative term: fused WMMA Gram + hinge --------------------------------
// grid.x = NGRP * 256 (256 M-tiles per group); block = 256 (8 waves).
// Each wave: fixed A tile (16 members), streams 224 B tiles, 2 per iteration
// so two independent WMMAs are in flight to hide latency/hazard NOPs.
__global__ void __launch_bounds__(256)
k_negative(const float* __restrict__ p2,
           const _Float16* __restrict__ fh,
           float* __restrict__ acc) {
  const int g    = blockIdx.x >> 8;
  const int mt   = blockIdx.x & 255;
  const int tid  = threadIdx.x;
  const int lane = tid & 31;
  const int wave = tid >> 5;
  const int half = lane >> 4;       // 0 | 1
  const int ln   = lane & 15;
  const int mbase = g * GM + mt * 16;

  // A fragment: lane holds row (mbase+ln), k = half*8+{0..7} and 16+half*8+{0..7}
  Frag A;
  const uint4* arow = reinterpret_cast<const uint4*>(fh + (size_t)(mbase + ln) * DIM);
  A.q[0] = arow[half];
  A.q[1] = arow[2 + half];

  float m2v[8], om2[8], om2h[8];
  #pragma unroll
  for (int r = 0; r < 8; ++r) {     // C/D reg r -> member mbase + r + 8*half
    const float v = p2[mbase + r + 8 * half];
    m2v[r]  = v;
    om2[r]  = 1.0f - v;
    om2h[r] = om2[r] * HALF_CM1;
  }

  float hinge = 0.0f;
  // 1792 negative tiles (2048 global row-tiles minus this group's 256);
  // wave handles idx = wave + 8j, two j's per iteration (1792/16 = 112 iters).
  for (int idx = wave; idx < 1792; idx += 16) {
    const int idx1 = idx + 8;
    const int nt0  = (idx  < g * 256) ? idx  : idx  + 256;
    const int nt1  = (idx1 < g * 256) ? idx1 : idx1 + 256;
    const int nb0  = nt0 * 16;
    const int nb1  = nt1 * 16;

    Frag B0, B1;
    const uint4* b0 = reinterpret_cast<const uint4*>(fh + (size_t)(nb0 + ln) * DIM);
    const uint4* b1 = reinterpret_cast<const uint4*>(fh + (size_t)(nb1 + ln) * DIM);
    B0.q[0] = b0[half];  B0.q[1] = b0[2 + half];
    B1.q[0] = b1[half];  B1.q[1] = b1[2 + half];
    const float n2a = p2[nb0 + ln];
    const float n2b = p2[nb1 + ln];

    v8f c0 = {};
    c0 = __builtin_amdgcn_wmma_f32_16x16x32_f16(false, A.v, false, B0.v,
                                                (short)0, c0, false, false);
    v8f c1 = {};
    c1 = __builtin_amdgcn_wmma_f32_16x16x32_f16(false, A.v, false, B1.v,
                                                (short)0, c1, false, false);

    hinge += hinge_tile(c0, n2a, m2v, om2, om2h);
    hinge += hinge_tile(c1, n2b, m2v, om2, om2h);
  }

  __shared__ float sdata[256];
  sdata[tid] = hinge;
  __syncthreads();
  #pragma unroll
  for (int s = 128; s > 0; s >>= 1) {
    if (tid < s) sdata[tid] += sdata[tid + s];
    __syncthreads();
  }
  if (tid == 0) atomicAdd(&acc[1], sdata[0]);
}

// ---- finalize: combine terms ------------------------------------------------
__global__ void k_finalize(const float* __restrict__ acc,
                           const int* __restrict__ kptr,
                           float* __restrict__ out) {
  if (threadIdx.x == 0) {
    int k = kptr[0];
    if (k <= 0 || k >= NNEG) k = NNEG;
    const float pos = acc[0] / (float)(NGRP * GM);
    const float neg = acc[1] / ((float)(NGRP * GM) * (float)k);
    out[0] = pos + neg;
  }
}

extern "C" void kernel_launch(void* const* d_in, const int* in_sizes, int n_in,
                              void* d_out, int out_size, void* d_ws, size_t ws_size,
                              hipStream_t stream) {
  (void)in_sizes; (void)n_in; (void)out_size; (void)ws_size;
  const float* emb  = (const float*)d_in[0];
  const int*   kptr = (const int*)d_in[3];

  char* ws = (char*)d_ws;
  float*    acc  = (float*)(ws + WS_ACC);
  float*    proj = (float*)(ws + WS_PROJ);
  float*    p2   = (float*)(ws + WS_P2);
  _Float16* fh   = (_Float16*)(ws + WS_FH);
  float*    cent = (float*)(ws + WS_CENT);
  float*    c2   = (float*)(ws + WS_C2);
  float*    out  = (float*)d_out;

  k_init<<<1, 32, 0, stream>>>(acc);
  k_project<<<NROWS / 8, 256, 0, stream>>>(emb, proj, p2, fh);
  k_centroid<<<NGRP, 256, 0, stream>>>(proj, cent, c2);
  k_positive<<<(NGRP * GM) / 8, 256, 0, stream>>>(proj, p2, cent, c2, acc);
  k_negative<<<NGRP * 256, 256, 0, stream>>>(p2, fh, acc);
  k_finalize<<<1, 32, 0, stream>>>(acc, kptr, out);
}
